// DDSPDecoder_87067577024580
// MI455X (gfx1250) — matrix operand order repaired and checked by reading, not compiled, not run on hardware
//
#include <hip/hip_runtime.h>
#include <hip/hip_bf16.h>
#include <math.h>

typedef __attribute__((ext_vector_type(16))) _Float16 v16h;
typedef __attribute__((ext_vector_type(8)))  _Float16 v8h;
typedef __attribute__((ext_vector_type(8)))  float    v8f;

#define BATCH   8
#define TFRM    500
#define DIN     128
#define HOPSZ   176
#define TAUD    88000
#define NPART   64
#define NBINS   256
#define SRATE   44100.0f
#define F_LO    5.0312195f      /* log2(32.7)  */
#define F_HI    11.0313520f     /* log2(2093)  */
#define N2      131072          /* pow2 FFT size >= T_audio */
#define N2H     (N2/2)
#define NFREQ2  (N2/2+1)
#define SCAN_B  256
#define NBLK    ((TAUD + SCAN_B - 1)/SCAN_B)   /* 344 */

// ---------------------------------------------------------------- utilities
__global__ void k_f32_to_f16(const float* __restrict__ src,
                             _Float16* __restrict__ dst, int n) {
  int i = blockIdx.x * blockDim.x + threadIdx.x;
  if (i < n) dst[i] = (_Float16)src[i];
}

// ------------------------------------------------------- WMMA tiled GEMM
// C[M,N] = act(A[M,K] @ B[K,N] + bias).  act==0: relu -> f16 out, act==1:
// sigmoid -> f32 out.
// Block: 8 waves; tile = 32 rows x 128 cols; each wave owns two 16x16 C
// tiles (rows r and r+16) sharing one B fragment -> 2 wmma per K-step.
// B tile kept column-major in LDS (stride 40 halfs) so both A and B
// fragments are fetched with ds_load_b128 pairs in the exact CDNA5 WMMA
// VGPR layouts.
__global__ void __launch_bounds__(256)
k_wmma_gemm(const _Float16* __restrict__ A, const _Float16* __restrict__ Bm,
            const float* __restrict__ bias, _Float16* __restrict__ outH,
            float* __restrict__ outF, int M, int N, int K, int act) {
  __shared__ _Float16 ldsA[32 * 32];        // 32 rows x 32 K, row-major
  __shared__ _Float16 ldsB[128 * 40];       // 128 cols x 32 K, col-major
  const int BSTR = 40;                      // halfs per column (pad 8)

  const int tid  = threadIdx.x;
  const int lane = tid & 31;
  const int wave = tid >> 5;
  const int mbase   = blockIdx.x * 32;
  const int colBase = blockIdx.y * 128;
  const int colG0   = colBase + wave * 16;
  const bool waveActive = (colG0 < N);      // uniform per wave -> EXEC all-1s

  const int r     = lane & 15;              // A row within 16-row subtile
  const int khalf = lane >> 4;              // K-half select
  const int cc    = wave * 16 + (lane & 15);// B column within LDS tile

  v8f acc0 = {};
  v8f acc1 = {};

  for (int k0 = 0; k0 < K; k0 += 32) {
    // ---- cooperative A tile load: 32x32 halfs, 4 halfs (b64) per thread
    {
      int idx = tid * 4;                    // 0..1020
      int row = idx >> 5;
      int c   = idx & 31;
      const uint2* gp =
          (const uint2*)(A + (size_t)(mbase + row) * K + k0 + c);
      *(uint2*)&ldsA[idx] = *gp;            // ds_store_b64
    }
    // ---- cooperative B tile load: 32 K x 128 cols, transposed into LDS
    {
      int k  = tid >> 3;                    // 0..31
      int cg = (tid & 7) * 16;              // 0..112
      int gc = colBase + cg;
      const _Float16* gB = Bm + (size_t)(k0 + k) * N + gc;
      _Float16 t[16];
      if (gc + 15 < N) {
        v8h lo = *(const v8h*)(gB);
        v8h hi = *(const v8h*)(gB + 8);
#pragma unroll
        for (int c = 0; c < 8; ++c) { t[c] = lo[c]; t[8 + c] = hi[c]; }
      } else {
#pragma unroll
        for (int c = 0; c < 16; ++c)
          t[c] = (gc + c < N) ? gB[c] : (_Float16)0.0f;
      }
#pragma unroll
      for (int c = 0; c < 16; ++c)          // col-major scatter
        ldsB[(cg + c) * BSTR + k] = t[c];
    }
    if (k0 + 32 < K)                        // hint next K tile
      __builtin_prefetch(A + (size_t)mbase * K + k0 + 32, 0, 1);
    __syncthreads();

    if (waveActive) {
      // B fragment: lane holds col cc, K-chunks [khalf*8..+7], [16+khalf*8..+7]
      v8h blo = *(const v8h*)&ldsB[cc * BSTR + khalf * 8];
      v8h bhi = *(const v8h*)&ldsB[cc * BSTR + 16 + khalf * 8];
      v16h bf;
#pragma unroll
      for (int i = 0; i < 8; ++i) { bf[i] = blo[i]; bf[8 + i] = bhi[i]; }

      // A fragments (rows r and r+16): K-chunks [khalf*8..+7], [16+khalf*8..+7]
      v8h a0lo = *(const v8h*)&ldsA[r * 32 + khalf * 8];
      v8h a0hi = *(const v8h*)&ldsA[r * 32 + 16 + khalf * 8];
      v8h a1lo = *(const v8h*)&ldsA[(r + 16) * 32 + khalf * 8];
      v8h a1hi = *(const v8h*)&ldsA[(r + 16) * 32 + 16 + khalf * 8];
      v16h af0, af1;
#pragma unroll
      for (int i = 0; i < 8; ++i) {
        af0[i] = a0lo[i]; af0[8 + i] = a0hi[i];
        af1[i] = a1lo[i]; af1[8 + i] = a1hi[i];
      }
      acc0 = __builtin_amdgcn_wmma_f32_16x16x32_f16(
          false, af0, false, bf, (short)0, acc0, false, false);
      acc1 = __builtin_amdgcn_wmma_f32_16x16x32_f16(
          false, af1, false, bf, (short)0, acc1, false, false);
    }
    __syncthreads();
  }

  if (waveActive) {
    int col = colG0 + (lane & 15);
    float bv = bias[col];
    int rowBase = mbase + (lane >> 4) * 8;  // C layout: VGPR i -> row (lane/16)*8+i
#pragma unroll
    for (int i = 0; i < 8; ++i) {
      float v0 = acc0[i] + bv;
      float v1 = acc1[i] + bv;
      if (act == 0) {
        outH[(size_t)(rowBase + i) * N + col]      = (_Float16)fmaxf(v0, 0.0f);
        outH[(size_t)(rowBase + i + 16) * N + col] = (_Float16)fmaxf(v1, 0.0f);
      } else {
        outF[(size_t)(rowBase + i) * N + col] =
            __builtin_amdgcn_rcpf(1.0f + __expf(-v0));
        outF[(size_t)(rowBase + i + 16) * N + col] =
            __builtin_amdgcn_rcpf(1.0f + __expf(-v1));
      }
    }
  }
}

// ---------------------------------------------------------------- f0 in Hz
__global__ void k_f0hz(const float* __restrict__ f0n,
                       float* __restrict__ f0hz, int n) {
  int i = blockIdx.x * blockDim.x + threadIdx.x;
  if (i < n) f0hz[i] = exp2f(f0n[i] * (F_HI - F_LO) + F_LO);
}

// -------------------------------------------- linear upsample f0 + loudness
__global__ void k_upsample2(const float* __restrict__ f0hz,
                            const float* __restrict__ loud,
                            float* __restrict__ f0up,
                            float* __restrict__ env) {
  int s = blockIdx.x * blockDim.x + threadIdx.x;
  int b = blockIdx.y;
  if (s >= TAUD) return;
  float src = (s + 0.5f) * ((float)TFRM / (float)TAUD) - 0.5f;
  src = fminf(fmaxf(src, 0.0f), (float)(TFRM - 1));
  int i0 = (int)src;
  int i1 = min(i0 + 1, TFRM - 1);
  float w = src - (float)i0;
  const float* fb = f0hz + b * TFRM;
  const float* lb = loud + b * TFRM;
  f0up[b * TAUD + s] = fb[i0] * (1.0f - w) + fb[i1] * w;
  env [b * TAUD + s] = lb[i0] * (1.0f - w) + lb[i1] * w;
}

// ------------------------------------------------------ blocked prefix scan
__global__ void k_scan1(const float* __restrict__ in, float* __restrict__ part,
                        float* __restrict__ bsum) {
  __shared__ float sh[SCAN_B];
  int b = blockIdx.y, blk = blockIdx.x, tid = threadIdx.x;
  int s = blk * SCAN_B + tid;
  float v = (s < TAUD) ? in[b * TAUD + s] : 0.0f;
  sh[tid] = v;
  __syncthreads();
  for (int off = 1; off < SCAN_B; off <<= 1) {
    float t = (tid >= off) ? sh[tid - off] : 0.0f;
    __syncthreads();
    sh[tid] += t;
    __syncthreads();
  }
  if (s < TAUD) part[b * TAUD + s] = sh[tid];
  if (tid == SCAN_B - 1) bsum[b * NBLK + blk] = sh[tid];
}

__global__ void k_scan2(float* __restrict__ bsum) {
  __shared__ float sh[512];
  int b = blockIdx.x, tid = threadIdx.x;
  float v = (tid < NBLK) ? bsum[b * NBLK + tid] : 0.0f;
  sh[tid] = v;
  __syncthreads();
  for (int off = 1; off < 512; off <<= 1) {
    float t = (tid >= off) ? sh[tid - off] : 0.0f;
    __syncthreads();
    sh[tid] += t;
    __syncthreads();
  }
  if (tid < NBLK) bsum[b * NBLK + tid] = sh[tid];
}

__global__ void k_scan3(const float* __restrict__ part,
                        const float* __restrict__ bsum,
                        float* __restrict__ phase) {
  int b = blockIdx.y, blk = blockIdx.x, tid = threadIdx.x;
  int s = blk * SCAN_B + tid;
  if (s >= TAUD) return;
  float off = (blk > 0) ? bsum[b * NBLK + blk - 1] : 0.0f;
  const float TWO_PI_SR = 6.283185307f / SRATE;
  phase[b * TAUD + s] = (part[b * TAUD + s] + off) * TWO_PI_SR;
}

// -------------------------------------------------------- harmonic synthesis
__global__ void k_harm(const float* __restrict__ phase,
                       const float* __restrict__ amp,
                       const float* __restrict__ f0hz,
                       float* __restrict__ harm) {
  int s = blockIdx.x * blockDim.x + threadIdx.x;
  int b = blockIdx.y;
  if (s >= TAUD) return;
  float ph = phase[b * TAUD + s];
  float src = (s + 0.5f) * ((float)TFRM / (float)TAUD) - 0.5f;
  src = fminf(fmaxf(src, 0.0f), (float)(TFRM - 1));
  int i0 = (int)src;
  int i1 = min(i0 + 1, TFRM - 1);
  float w = src - (float)i0;
  int tn = min(s / HOPSZ, TFRM - 1);
  float f0 = f0hz[b * TFRM + tn];
  const float* a0 = amp + (size_t)(b * TFRM + i0) * NPART;
  const float* a1 = amp + (size_t)(b * TFRM + i1) * NPART;
  float acc = 0.0f;
#pragma unroll 4
  for (int p = 1; p <= NPART; ++p) {
    float fa = a0[p - 1] * (1.0f - w) + a1[p - 1] * w;
    float m = (f0 * (float)p < SRATE * 0.5f) ? 1.0f : 0.0f;
    acc += __sinf(ph * (float)p) * fa * m;
  }
  harm[b * TAUD + s] = acc;
}

// ----------------------- exact per-frame weights of mean(lin_interp(x, Ta))
__global__ void k_coeff(float* __restrict__ coeff) {
  int t = blockIdx.x * blockDim.x + threadIdx.x;
  if (t >= TFRM) return;
  float acc = 0.0f;
  int s0 = max(0, (t - 2) * HOPSZ);
  int s1 = min(TAUD, (t + 3) * HOPSZ);
  for (int s = s0; s < s1; ++s) {
    float src = (s + 0.5f) * ((float)TFRM / (float)TAUD) - 0.5f;
    src = fminf(fmaxf(src, 0.0f), (float)(TFRM - 1));
    int i0 = (int)src;
    int i1 = min(i0 + 1, TFRM - 1);
    float w = src - (float)i0;
    if (i0 == t) acc += 1.0f - w;
    if (i1 == t) acc += w;
  }
  coeff[t] = acc / (float)TAUD;
}

__global__ void k_meanmags(const float* __restrict__ nmag,
                           const float* __restrict__ coeff,
                           float* __restrict__ mm) {
  int b = blockIdx.x, c = threadIdx.x;
  float acc = 0.0f;
  for (int t = 0; t < TFRM; ++t)
    acc += coeff[t] * nmag[(size_t)(b * TFRM + t) * NBINS + c];
  mm[b * NBINS + c] = acc;
}

// -------------------------------------- filter resampled to FFT bin grid
__global__ void k_filt(const float* __restrict__ mm, float* __restrict__ filt) {
  int k = blockIdx.x * blockDim.x + threadIdx.x;
  int b = blockIdx.y;
  if (k >= NFREQ2) return;
  float src = (k + 0.5f) * ((float)NBINS / (float)NFREQ2) - 0.5f;
  src = fminf(fmaxf(src, 0.0f), (float)(NBINS - 1));
  int i0 = (int)src;
  int i1 = min(i0 + 1, NBINS - 1);
  float w = src - (float)i0;
  const float* m = mm + b * NBINS;
  filt[(size_t)b * NFREQ2 + k] = m[i0] * (1.0f - w) + m[i1] * w;
}

// ---------------------------------------------------- Stockham radix-2 FFT
__global__ void k_fft_pack(const float* __restrict__ noise,
                           float2* __restrict__ X) {
  int i = blockIdx.x * blockDim.x + threadIdx.x;
  int b = blockIdx.y;
  if (i >= N2) return;
  float2 v;
  v.x = (i < TAUD) ? noise[b * TAUD + i] : 0.0f;
  v.y = 0.0f;
  X[(size_t)b * N2 + i] = v;
}

__global__ void k_fft_stage(const float2* __restrict__ X,
                            float2* __restrict__ Y, int Ns, float sign) {
  int j = blockIdx.x * blockDim.x + threadIdx.x;
  int b = blockIdx.y;
  if (j >= N2H) return;
  const float2* Xb = X + (size_t)b * N2;
  float2 a = Xb[j];
  float2 c = Xb[j + N2H];
  int jm = j & (Ns - 1);
  float ang = sign * 3.14159265359f * (float)jm / (float)Ns;  // 2*pi*jm/(2*Ns)
  float sn, cs;
  __sincosf(ang, &sn, &cs);
  float2 cw;
  cw.x = c.x * cs - c.y * sn;
  cw.y = c.x * sn + c.y * cs;
  int out0 = ((j & ~(Ns - 1)) << 1) | jm;
  float2 y0, y1;
  y0.x = a.x + cw.x; y0.y = a.y + cw.y;
  y1.x = a.x - cw.x; y1.y = a.y - cw.y;
  Y[(size_t)b * N2 + out0]      = y0;
  Y[(size_t)b * N2 + out0 + Ns] = y1;
}

__global__ void k_fft_filt(float2* __restrict__ X,
                           const float* __restrict__ filt) {
  int k = blockIdx.x * blockDim.x + threadIdx.x;
  int b = blockIdx.y;
  if (k >= N2) return;
  int kk = (k <= N2H) ? k : (N2 - k);   // hermitian-symmetric real filter
  float f = filt[(size_t)b * NFREQ2 + kk];
  float2 v = X[(size_t)b * N2 + k];
  v.x *= f; v.y *= f;
  X[(size_t)b * N2 + k] = v;
}

__global__ void k_fft_unpack(const float2* __restrict__ X,
                             float* __restrict__ nz) {
  int s = blockIdx.x * blockDim.x + threadIdx.x;
  int b = blockIdx.y;
  if (s >= TAUD) return;
  nz[b * TAUD + s] = X[(size_t)b * N2 + s].x * (0.1f / (float)N2);
}

// ----------------------------------------------------------------- combine
__global__ void k_combine(const float* __restrict__ harm,
                          const float* __restrict__ nz,
                          const float* __restrict__ env,
                          float* __restrict__ out, int n) {
  int i = blockIdx.x * blockDim.x + threadIdx.x;
  if (i < n) out[i] = (harm[i] + nz[i]) * env[i];
}

// =========================================================== host launcher
extern "C" void kernel_launch(void* const* d_in, const int* in_sizes, int n_in,
                              void* d_out, int out_size, void* d_ws,
                              size_t ws_size, hipStream_t stream) {
  const float* z    = (const float*)d_in[0];
  const float* f0n  = (const float*)d_in[1];
  const float* loud = (const float*)d_in[2];
  const float* noise= (const float*)d_in[3];
  const float* W1p  = (const float*)d_in[4];
  const float* b1p  = (const float*)d_in[5];
  const float* W2p  = (const float*)d_in[6];
  const float* b2p  = (const float*)d_in[7];
  const float* W1n  = (const float*)d_in[8];
  const float* b1n  = (const float*)d_in[9];
  const float* W2n  = (const float*)d_in[10];
  const float* b2n  = (const float*)d_in[11];
  float* out = (float*)d_out;

  char* p = (char*)d_ws;
  auto alloc = [&](size_t bytes) -> void* {
    char* r = p;
    p += (bytes + 255) & ~(size_t)255;
    return (void*)r;
  };

  const int MROWS = BATCH * TFRM;  // 4000
  _Float16* zh   = (_Float16*)alloc((size_t)MROWS * DIN * 2);
  _Float16* w1ph = (_Float16*)alloc((size_t)DIN * 256 * 2);
  _Float16* w2ph = (_Float16*)alloc((size_t)256 * NPART * 2);
  _Float16* w1nh = (_Float16*)alloc((size_t)DIN * 256 * 2);
  _Float16* w2nh = (_Float16*)alloc((size_t)256 * NBINS * 2);
  _Float16* Hp   = (_Float16*)alloc((size_t)MROWS * 256 * 2);
  _Float16* Hn   = (_Float16*)alloc((size_t)MROWS * 256 * 2);
  float* amp    = (float*)alloc((size_t)MROWS * NPART * 4);
  float* nmag   = (float*)alloc((size_t)MROWS * NBINS * 4);
  float* f0hz   = (float*)alloc((size_t)MROWS * 4);
  float* f0up   = (float*)alloc((size_t)BATCH * TAUD * 4);
  float* env    = (float*)alloc((size_t)BATCH * TAUD * 4);
  float* phase  = (float*)alloc((size_t)BATCH * TAUD * 4);
  float* harm   = (float*)alloc((size_t)BATCH * TAUD * 4);
  float* nzout  = (float*)alloc((size_t)BATCH * TAUD * 4);
  float* bsum   = (float*)alloc((size_t)BATCH * NBLK * 4);
  float* coeff  = (float*)alloc((size_t)TFRM * 4);
  float* mmags  = (float*)alloc((size_t)BATCH * NBINS * 4);
  float* filt   = (float*)alloc((size_t)BATCH * NFREQ2 * 4);
  float2* fftA  = (float2*)alloc((size_t)BATCH * N2 * 8);
  float2* fftB  = (float2*)alloc((size_t)BATCH * N2 * 8);

  // ---- precision conversion for WMMA path
  k_f32_to_f16<<<(MROWS * DIN + 255) / 256, 256, 0, stream>>>(z, zh, MROWS * DIN);
  k_f32_to_f16<<<(DIN * 256 + 255) / 256, 256, 0, stream>>>(W1p, w1ph, DIN * 256);
  k_f32_to_f16<<<(256 * NPART + 255) / 256, 256, 0, stream>>>(W2p, w2ph, 256 * NPART);
  k_f32_to_f16<<<(DIN * 256 + 255) / 256, 256, 0, stream>>>(W1n, w1nh, DIN * 256);
  k_f32_to_f16<<<(256 * NBINS + 255) / 256, 256, 0, stream>>>(W2n, w2nh, 256 * NBINS);

  // ---- MLP heads on WMMA (32-row x 128-col blocks)
  k_wmma_gemm<<<dim3(MROWS / 32, 2), 256, 0, stream>>>(
      zh, w1ph, b1p, Hp, nullptr, MROWS, 256, DIN, 0);
  k_wmma_gemm<<<dim3(MROWS / 32, 1), 256, 0, stream>>>(
      Hp, w2ph, b2p, nullptr, amp, MROWS, NPART, 256, 1);
  k_wmma_gemm<<<dim3(MROWS / 32, 2), 256, 0, stream>>>(
      zh, w1nh, b1n, Hn, nullptr, MROWS, 256, DIN, 0);
  k_wmma_gemm<<<dim3(MROWS / 32, 2), 256, 0, stream>>>(
      Hn, w2nh, b2n, nullptr, nmag, MROWS, NBINS, 256, 1);

  // ---- harmonic branch
  k_f0hz<<<(MROWS + 255) / 256, 256, 0, stream>>>(f0n, f0hz, MROWS);
  k_upsample2<<<dim3(NBLK, BATCH), 256, 0, stream>>>(f0hz, loud, f0up, env);
  k_scan1<<<dim3(NBLK, BATCH), SCAN_B, 0, stream>>>(f0up, phase, bsum);
  k_scan2<<<BATCH, 512, 0, stream>>>(bsum);
  k_scan3<<<dim3(NBLK, BATCH), SCAN_B, 0, stream>>>(phase, bsum, phase);
  k_harm<<<dim3(NBLK, BATCH), 256, 0, stream>>>(phase, amp, f0hz, harm);

  // ---- noise branch
  k_coeff<<<(TFRM + 255) / 256, 256, 0, stream>>>(coeff);
  k_meanmags<<<BATCH, NBINS, 0, stream>>>(nmag, coeff, mmags);
  k_filt<<<dim3((NFREQ2 + 255) / 256, BATCH), 256, 0, stream>>>(mmags, filt);

  k_fft_pack<<<dim3(N2 / 256, BATCH), 256, 0, stream>>>(noise, fftA);
  float2* cur = fftA;
  float2* oth = fftB;
  for (int Ns = 1; Ns < N2; Ns <<= 1) {
    k_fft_stage<<<dim3(N2H / 256, BATCH), 256, 0, stream>>>(cur, oth, Ns, -1.0f);
    float2* t = cur; cur = oth; oth = t;
  }
  k_fft_filt<<<dim3(N2 / 256, BATCH), 256, 0, stream>>>(cur, filt);
  for (int Ns = 1; Ns < N2; Ns <<= 1) {
    k_fft_stage<<<dim3(N2H / 256, BATCH), 256, 0, stream>>>(cur, oth, Ns, 1.0f);
    float2* t = cur; cur = oth; oth = t;
  }
  k_fft_unpack<<<dim3(NBLK, BATCH), 256, 0, stream>>>(cur, nzout);

  // ---- combine
  k_combine<<<(BATCH * TAUD + 255) / 256, 256, 0, stream>>>(
      harm, nzout, env, out, BATCH * TAUD);
}
